// SelfAttention_1812476198915
// MI455X (gfx1250) — matrix operand order repaired and checked by reading, not compiled, not run on hardware
//
#include <hip/hip_runtime.h>
#include <hip/hip_bf16.h>
#include <cstdint>

// Problem sizes (fixed by the reference)
#define BB 4
#define CC 256
#define NN 4096   // H*W = 64*64
#define DD 32     // C/8
#define NCHUNKS (NN / 32)
#define VROW 40   // padded LDS row stride in halfwords: 32 data + 8 pad (80B)

typedef _Float16 v8h  __attribute__((ext_vector_type(8)));
typedef _Float16 v16h __attribute__((ext_vector_type(16)));
typedef float    v8f  __attribute__((ext_vector_type(8)));
typedef unsigned int u32x4 __attribute__((ext_vector_type(4)));
typedef int          i32x8 __attribute__((ext_vector_type(8)));
typedef int          i32x4 __attribute__((ext_vector_type(4)));

#if defined(__has_builtin)
#if __has_builtin(__builtin_amdgcn_tensor_load_to_lds) && \
    __has_builtin(__builtin_amdgcn_s_wait_tensorcnt)
#define USE_TDM 1
#endif
#endif

static __device__ __forceinline__ v8f wmma_f32_f16(v16h a, v16h b, v8f c) {
  // D = A(16x32 f16) * B(32x16 f16) + C(16x16 f32)
  return __builtin_amdgcn_wmma_f32_16x16x32_f16(false, a, false, b, (short)0, c,
                                                false, false);
}

// A-operand (16x32 f16, M=row, K over 32): lane (g = lane>>4, row = lane&15)
// element j -> K = 16g + j. With [row][32] row-major storage this is
// two contiguous 16B chunks at halfwords (16g) and (16g+8).
static __device__ __forceinline__ v16h load_a_row(const _Float16* base, int row, int g) {
  const _Float16* p = base + (size_t)row * DD;
  v8h lo = *(const v8h*)(p + 16 * g);
  v8h hi = *(const v8h*)(p + 16 * g + 8);
  return __builtin_shufflevector(lo, hi, 0, 1, 2, 3, 4, 5, 6, 7,
                                 8, 9, 10, 11, 12, 13, 14, 15);
}

#ifdef USE_TDM
// Tensor Data Mover: DMA a [256 rows x 32 cols] f16 tile of V (row stride NN)
// into LDS at lds_byte_off, padding 16B after every 64B row (bank-conflict-free
// ds_load of B-operands afterwards). D# per CDNA5 ISA.
// This toolchain exposes the 6-arg builtin:
//   (u32x4 g0, i32x8 g1, i32x4 g2, i32x4 g3, i32x8 extra, i32 cpol)
static __device__ __forceinline__ void tdm_load_v_tile(const _Float16* gsrc,
                                                       uint32_t lds_byte_off) {
  const uint64_t ga = (uint64_t)(uintptr_t)gsrc;
  u32x4 g0;
  g0[0] = 1u;                                   // count=1, user mode, no gather
  g0[1] = lds_byte_off;                         // lds_addr
  g0[2] = (uint32_t)ga;                         // global_addr[31:0]
  g0[3] = (uint32_t)(ga >> 32) | (2u << 30);    // global_addr[56:32], type=2
  i32x8 g1;
  g1[0] = (1 << 16)      // data_size = 2 bytes
        | (1 << 20)      // pad_enable
        | (3 << 22)      // pad_interval: 16 DWORDs (64B row)
        | (3 << 25);     // pad_amount: 4 DWORDs (16B)
  g1[1] = (int)(4096u << 16);   // tensor_dim0 = 4096 (bits 79:48, low half)
  g1[2] = (int)(256u << 16);    // tensor_dim0 hi = 0, tensor_dim1 = 256
  g1[3] = (int)(32u << 16);     // tensor_dim1 hi = 0, tile_dim0 = 32
  g1[4] = 256;                  // tile_dim1 = 256, tile_dim2 = 0
  g1[5] = 4096;                 // tensor_dim0_stride = 4096 elements
  g1[6] = 0;                    // stride0 hi, tensor_dim1_stride lo
  g1[7] = 0;                    // tensor_dim1_stride hi
  const i32x4 gz4 = {0, 0, 0, 0};               // groups 2/3 unused (2D tensor)
  const i32x8 gz8 = {0, 0, 0, 0, 0, 0, 0, 0};
  __builtin_amdgcn_tensor_load_to_lds(g0, g1, gz4, gz4, gz8, 0);
}
#endif

// ---------------------------------------------------------------------------
// Kernel 1: q16[b][n][d] = Wq[d,:].x[:,n] + bq ; kt16[b][m][d] = Wk[d,:].x[:,m] + bk
// ---------------------------------------------------------------------------
__global__ void __launch_bounds__(64) proj_qk_kernel(
    const float* __restrict__ x,
    const float* __restrict__ Wq, const float* __restrict__ bq,
    const float* __restrict__ Wk, const float* __restrict__ bk,
    _Float16* __restrict__ q16, _Float16* __restrict__ kt16) {
  const int gid = blockIdx.x * 64 + threadIdx.x;  // 0 .. B*N-1
  const int b = gid >> 12;                        // / 4096
  const int n = gid & (NN - 1);
  const float* xb = x + (size_t)b * CC * NN + n;

  float accq[DD], acck[DD];
#pragma unroll
  for (int d = 0; d < DD; ++d) { accq[d] = bq[d]; acck[d] = bk[d]; }

  for (int c = 0; c < CC; ++c) {
    float xv = xb[(size_t)c * NN];               // coalesced across threads
#pragma unroll
    for (int d = 0; d < DD; ++d) {
      accq[d] = fmaf(Wq[d * CC + c], xv, accq[d]);   // uniform -> s_load
      acck[d] = fmaf(Wk[d * CC + c], xv, acck[d]);
    }
  }
  _Float16* qr = q16  + (size_t)gid * DD;
  _Float16* kr = kt16 + (size_t)gid * DD;
#pragma unroll
  for (int d = 0; d < DD; ++d) { qr[d] = (_Float16)accq[d]; kr[d] = (_Float16)acck[d]; }
}

// ---------------------------------------------------------------------------
// Kernel 2: v16[b][c][n] = Wv[c,:].x[:,n] + bv[c]   (f16, row-major over n)
// ---------------------------------------------------------------------------
__global__ void __launch_bounds__(256) proj_v_kernel(
    const float* __restrict__ x,
    const float* __restrict__ Wv, const float* __restrict__ bv,
    _Float16* __restrict__ v16) {
  // grid = BB * (NN/256) * (CC/32) = 512 blocks
  const int blk = blockIdx.x;
  const int b   = blk >> 7;
  const int nch = (blk >> 3) & 15;
  const int cch = blk & 7;
  const int n   = nch * 256 + threadIdx.x;
  const int c0  = cch * 32;
  const float* xb = x + (size_t)b * CC * NN + n;

  float acc[32];
#pragma unroll
  for (int i = 0; i < 32; ++i) acc[i] = bv[c0 + i];

  for (int c = 0; c < CC; ++c) {
    float xv = xb[(size_t)c * NN];
#pragma unroll
    for (int i = 0; i < 32; ++i)
      acc[i] = fmaf(Wv[(c0 + i) * CC + c], xv, acc[i]);
  }
#pragma unroll
  for (int i = 0; i < 32; ++i)
    v16[((size_t)b * CC + c0 + i) * NN + n] = (_Float16)acc[i];
}

// ---------------------------------------------------------------------------
// Kernel 3: per output column m: M[m] = max_n s[n,m], L[m] = sum_n exp(s-M)
// One wave per 16-wide m tile; S tiles via WMMA, online max/sum over n.
// ---------------------------------------------------------------------------
__global__ void __launch_bounds__(128) attn_stats_kernel(
    const _Float16* __restrict__ q16, const _Float16* __restrict__ kt16,
    float* __restrict__ Mstat, float* __restrict__ Lstat) {
  const int lane = threadIdx.x & 31;
  const int wid  = threadIdx.x >> 5;
  const int gmt  = blockIdx.x * 4 + wid;  // 0..1023
  const int b    = gmt >> 8;
  const int mt   = gmt & 255;
  const int lo = lane & 15, g = lane >> 4;

  const _Float16* qb = q16 + (size_t)b * NN * DD;
  // B-operand (Kt, 32x16): lane (g, m) element j -> d = 16g + j : contiguous 32B
  v16h bmat = *(const v16h*)(kt16 + ((size_t)b * NN + mt * 16 + lo) * DD + 16 * g);

  float rmax = -3.0e38f;
  float rsum = 0.0f;
  const v8f zero = {};
  for (int nb = 0; nb < NN; nb += 32) {
    v16h a0 = load_a_row(qb, nb + lo, g);
    v16h a1 = load_a_row(qb, nb + 16 + lo, g);
    v8f s0 = wmma_f32_f16(a0, bmat, zero);
    v8f s1 = wmma_f32_f16(a1, bmat, zero);
    float t = s0[0];
#pragma unroll
    for (int r = 1; r < 8; ++r) t = fmaxf(t, s0[r]);
#pragma unroll
    for (int r = 0; r < 8; ++r) t = fmaxf(t, s1[r]);
    float nm = fmaxf(rmax, t);
    float sc = __expf(rmax - nm);
    float ls = 0.0f;
#pragma unroll
    for (int r = 0; r < 8; ++r) ls += __expf(s0[r] - nm) + __expf(s1[r] - nm);
    rsum = fmaf(rsum, sc, ls);
    rmax = nm;
  }
  // lanes l and l^16 hold complementary row halves of the same column
  float omax = __shfl_xor(rmax, 16, 32);
  float osum = __shfl_xor(rsum, 16, 32);
  float M = fmaxf(rmax, omax);
  float S = osum * __expf(omax - M) + rsum * __expf(rmax - M);
  if (g == 0) {
    Mstat[(size_t)b * NN + mt * 16 + lo] = M;
    Lstat[(size_t)b * NN + mt * 16 + lo] = S;
  }
}

// ---------------------------------------------------------------------------
// Kernel 4: outT[m,c] = sum_n p[n,m] * v[c,n]; out = gamma*(outT/L) + x
// 4 waves/block (one 16-m tile each, same batch & n-schedule) share the
// [256 x 32] V tile, staged into LDS by the Tensor Data Mover (double-
// buffered, s_wait_tensorcnt), with D# row padding for conflict-free ds_loads.
// exp(S)-tile -> Pt A-operand is a pure in-lane f32->f16 repack.
// ---------------------------------------------------------------------------
__global__ void __launch_bounds__(128) attn_out_kernel(
    const _Float16* __restrict__ q16, const _Float16* __restrict__ kt16,
    const _Float16* __restrict__ v16,
    const float* __restrict__ Mstat, const float* __restrict__ Lstat,
    const float* __restrict__ x, const float* __restrict__ gamma,
    float* __restrict__ out) {
  __shared__ __align__(64) _Float16 vtile[2][256 * VROW];  // 2 x 20KB

  const int lane = threadIdx.x & 31;
  const int wid  = threadIdx.x >> 5;
  const int gmt  = blockIdx.x * 4 + wid;  // 0..1023 (batch uniform per block)
  const int b    = gmt >> 8;
  const int mt   = gmt & 255;
  const int lo = lane & 15, g = lane >> 4;

  const _Float16* qb = q16 + (size_t)b * NN * DD;
  const _Float16* vb = v16 + (size_t)b * CC * NN;
  v16h bmat = *(const v16h*)(kt16 + ((size_t)b * NN + mt * 16 + lo) * DD + 16 * g);

  const float Mcol = Mstat[(size_t)b * NN + mt * 16 + lo];
  float linv[8];
#pragma unroll
  for (int r = 0; r < 8; ++r)
    linv[r] = 1.0f / Lstat[(size_t)b * NN + mt * 16 + r + 8 * g];

  const v8f zero = {};
  v8f acc[16];
#pragma unroll
  for (int t = 0; t < 16; ++t) acc[t] = zero;

#ifdef USE_TDM
  const uint32_t lds0 = (uint32_t)(uintptr_t)&vtile[0][0];
  const uint32_t lds1 = (uint32_t)(uintptr_t)&vtile[1][0];
  if (wid == 0) tdm_load_v_tile(vb, lds0);  // prefetch chunk 0
#endif

  for (int it = 0; it < NCHUNKS; ++it) {
    const int nb  = it * 32;
    const int cur = it & 1;
#ifdef USE_TDM
    if (wid == 0) {
      if (it + 1 < NCHUNKS) {
        tdm_load_v_tile(vb + (size_t)(it + 1) * 32, cur ? lds0 : lds1);
        __builtin_amdgcn_s_wait_tensorcnt(1);   // chunk `it` landed
      } else {
        __builtin_amdgcn_s_wait_tensorcnt(0);
      }
    }
    __syncthreads();                            // publish vtile[cur]
#else
    __syncthreads();
    for (int r = threadIdx.x; r < CC; r += 128) {
      const _Float16* src = vb + (size_t)r * NN + nb;
      _Float16* dst = &vtile[cur][r * VROW];
#pragma unroll
      for (int j = 0; j < 4; ++j)
        *(v8h*)(dst + 8 * j) = *(const v8h*)(src + 8 * j);
    }
    __syncthreads();
#endif

    v16h a0 = load_a_row(qb, nb + lo, g);
    v16h a1 = load_a_row(qb, nb + 16 + lo, g);
    v8f s0 = wmma_f32_f16(a0, bmat, zero);   // rows nb..nb+15
    v8f s1 = wmma_f32_f16(a1, bmat, zero);   // rows nb+16..nb+31
    // C-layout of s0/s1 maps in-lane onto the A-layout of Pt (K = n_local)
    v16h p;
#pragma unroll
    for (int j = 0; j < 8; ++j) {
      p[j]     = (_Float16)__expf(s0[j] - Mcol);
      p[j + 8] = (_Float16)__expf(s1[j] - Mcol);
    }
    const _Float16* vt = &vtile[cur][0];
#pragma unroll
    for (int t = 0; t < 16; ++t) {
      // B-operand = V^T (K=n, N=c): lane (g, c) element j -> v[c][nb+j+8g+8*(j>=8)],
      // 2 x 16B ds_loads from the padded LDS tile (conflict-free per phase).
      const int roff = (t * 16 + lo) * VROW + 8 * g;
      v8h blo = *(const v8h*)(vt + roff);
      v8h bhi = *(const v8h*)(vt + roff + 16);
      v16h bv = __builtin_shufflevector(blo, bhi, 0, 1, 2, 3, 4, 5, 6, 7,
                                        8, 9, 10, 11, 12, 13, 14, 15);
      acc[t] = wmma_f32_f16(p, bv, acc[t]);
    }
    __syncthreads();   // all waves done reading vtile[cur] before TDM reuses it
  }

  const float gam = gamma[0];
#pragma unroll
  for (int t = 0; t < 16; ++t) {
#pragma unroll
    for (int r = 0; r < 8; ++r) {
      const size_t c = (size_t)t * 16 + lo;           // C/D lane column = c
      const size_t m = (size_t)mt * 16 + r + 8 * g;   // C/D VGPR row    = m
      const size_t idx = ((size_t)b * CC + c) * NN + m;
      out[idx] = fmaf(gam, acc[t][r] * linv[r], x[idx]);
    }
  }
}

// ---------------------------------------------------------------------------
extern "C" void kernel_launch(void* const* d_in, const int* in_sizes, int n_in,
                              void* d_out, int out_size, void* d_ws, size_t ws_size,
                              hipStream_t stream) {
  (void)in_sizes; (void)n_in; (void)out_size; (void)ws_size;
  const float* x     = (const float*)d_in[0];
  const float* Wq    = (const float*)d_in[1];
  const float* bq    = (const float*)d_in[2];
  const float* Wk    = (const float*)d_in[3];
  const float* bk    = (const float*)d_in[4];
  const float* Wv    = (const float*)d_in[5];
  const float* bv    = (const float*)d_in[6];
  const float* gamma = (const float*)d_in[7];
  float* out = (float*)d_out;

  // Workspace layout (~10.6 MB)
  char* ws = (char*)d_ws;
  const size_t qk_bytes = (size_t)BB * NN * DD * sizeof(_Float16);   // 1 MB each
  const size_t v_bytes  = (size_t)BB * CC * NN * sizeof(_Float16);   // 8 MB
  _Float16* q16  = (_Float16*)(ws);
  _Float16* kt16 = (_Float16*)(ws + qk_bytes);
  _Float16* v16  = (_Float16*)(ws + 2 * qk_bytes);
  float* Mstat   = (float*)(ws + 2 * qk_bytes + v_bytes);
  float* Lstat   = (float*)(ws + 2 * qk_bytes + v_bytes + (size_t)BB * NN * sizeof(float));

  proj_qk_kernel<<<(BB * NN) / 64, 64, 0, stream>>>(x, Wq, bq, Wk, bk, q16, kt16);
  proj_v_kernel<<<BB * (NN / 256) * (CC / 32), 256, 0, stream>>>(x, Wv, bv, v16);
  attn_stats_kernel<<<(BB * (NN / 16)) / 4, 128, 0, stream>>>(q16, kt16, Mstat, Lstat);
  attn_out_kernel<<<(BB * (NN / 16)) / 4, 128, 0, stream>>>(q16, kt16, v16, Mstat, Lstat,
                                                            x, gamma, out);
}
